// MatchNet_22952305229978
// MI455X (gfx1250) — compile-verified, matrix operand order checked by reading, not run on hardware
//
#include <hip/hip_runtime.h>
#include <hip/hip_bf16.h>
#include <math.h>

// ---------------------------------------------------------------------------
// MatchNet forward: z = MLP(Xf); x = PDHG_LP_solve(S, b=Xf, z), 60 iters.
// Fused persistent kernel: one WG (8 wave32) owns 16 batch rows for all 60
// PDHG iterations. S (0/1, exact in f16) is staged in LDS in BOTH layouts
// (row-major and transposed, ~272 KB -- only possible with CDNA5's 320 KB
// WGP LDS). Both per-iteration GEMMs run on v_wmma_f32_16x16x32_f16 with
// f32 accumulation; all solver state stays in VGPRs in WMMA fragment layout.
// ---------------------------------------------------------------------------

#define N_COMBOS  128
#define N_STRUCTS 512
#define BATCHSZ   2048
#define HID       20
#define PDHG_ITERS 60
#define CONTROLF  10.0f

typedef _Float16 v16h __attribute__((ext_vector_type(16)));
typedef _Float16 v8h  __attribute__((ext_vector_type(8)));
typedef float    v8f  __attribute__((ext_vector_type(8)));

// LDS row pitches (f16 elems), padded so pitch*2B = 16B-aligned and
// pitch/2 dwords == 4 (mod 64 banks) -> 16 lanes hit 16 distinct banks.
#define SC_PITCH 520   // S  row-major  [combo][struct]
#define ST_PITCH 136   // S^T row-major [struct][combo]
#define XB_PITCH 520   // xb f16        [row][struct]
#define L1_PITCH 136   // lam1 f16      [row][combo]

// Load a 16-bit WMMA fragment: per the CDNA5 A layout, each lane holds two
// contiguous 8-halfword (16B) runs -> two ds_load_b128.
__device__ inline v16h ld_frag(const _Float16* p0, const _Float16* p1) {
  v8h a = *(const v8h*)p0;
  v8h b = *(const v8h*)p1;
  v16h r;
#pragma unroll
  for (int i = 0; i < 8; ++i) { r[i] = a[i]; r[i + 8] = b[i]; }
  return r;
}

// ---------------------------------------------------------------------------
// Kernel 0: spectral bound of [S; I] via 30 power iterations (constant work,
// one WG). Writes tau = sig = 0.9 / L to workspace.
// ---------------------------------------------------------------------------
__global__ __launch_bounds__(256) void spectral_kernel(
    const float* __restrict__ S, float* __restrict__ tau_out) {
  extern __shared__ __align__(16) char smem_raw[];
  float* Sl  = (float*)smem_raw;            // 128*512
  float* v   = Sl + N_COMBOS * N_STRUCTS;   // 512
  float* sv  = v + N_STRUCTS;               // 128
  float* red = sv + N_COMBOS;               // 256
  const int t = threadIdx.x;

  for (int i = t; i < N_COMBOS * N_STRUCTS; i += 256) Sl[i] = S[i];
  for (int i = t; i < N_STRUCTS; i += 256) v[i] = 1.0f;
  __syncthreads();

  for (int it = 0; it < 30; ++it) {
    if (t < N_COMBOS) {
      float a = 0.f;
      for (int n = 0; n < N_STRUCTS; ++n) a += Sl[t * N_STRUCTS + n] * v[n];
      sv[t] = a;
    }
    __syncthreads();
    // u = S^T (S v) + v   (each thread owns columns t and t+256)
    float a0 = v[t], a1 = v[t + 256];
    for (int m = 0; m < N_COMBOS; ++m) {
      float s = sv[m];
      a0 += Sl[m * N_STRUCTS + t] * s;
      a1 += Sl[m * N_STRUCTS + t + 256] * s;
    }
    red[t] = a0 * a0 + a1 * a1;
    __syncthreads();
    for (int w = 128; w > 0; w >>= 1) {
      if (t < w) red[t] += red[t + w];
      __syncthreads();
    }
    float inv = rsqrtf(red[0]);
    v[t] = a0 * inv;
    v[t + 256] = a1 * inv;
    __syncthreads();
  }
  // L^2 = ||S v||^2 + ||v||^2, with ||v|| == 1 after normalization
  if (t < N_COMBOS) {
    float a = 0.f;
    for (int n = 0; n < N_STRUCTS; ++n) a += Sl[t * N_STRUCTS + n] * v[n];
    sv[t] = a;
  }
  __syncthreads();
  red[t] = (t < N_COMBOS) ? sv[t] * sv[t] : 0.f;
  __syncthreads();
  for (int w = 128; w > 0; w >>= 1) {
    if (t < w) red[t] += red[t + w];
    __syncthreads();
  }
  if (t == 0) tau_out[0] = 0.9f / sqrtf(red[0] + 1.0f);
}

// ---------------------------------------------------------------------------
// Kernel 1: fused MLP + 60 PDHG iterations for a 16-row batch tile.
// 256 threads = 8 wave32. Wave w owns combo tile [16w,16w+16) for u1 and
// structure tiles {64w+16t} t=0..3 for p. All elementwise state is kept in
// registers in the WMMA C/D fragment layout:
//   lane l, acc elem r  <->  (m = r + 8*(l>>4), n = (l&15) + tile_base)
// ---------------------------------------------------------------------------
__global__ __launch_bounds__(256) void pdhg_kernel(
    const float* __restrict__ X,
    const float* __restrict__ W1, const float* __restrict__ Bb1,
    const float* __restrict__ W2, const float* __restrict__ Bb2,
    const float* __restrict__ W3, const float* __restrict__ Bb3,
    const float* __restrict__ W4, const float* __restrict__ Bb4,
    const float* __restrict__ Sg, const float* __restrict__ tau_ptr,
    float* __restrict__ out) {
  extern __shared__ __align__(16) char smem_raw[];
  _Float16* Sc  = (_Float16*)smem_raw;        // [128][SC_PITCH]  ~130 KB
  _Float16* St  = Sc  + N_COMBOS * SC_PITCH;  // [512][ST_PITCH]  ~136 KB
  _Float16* xbh = St  + N_STRUCTS * ST_PITCH; // [16][XB_PITCH]    ~16 KB
  _Float16* l1h = xbh + 16 * XB_PITCH;        // [16][L1_PITCH]     ~4 KB
  float* xf  = (float*)(l1h + 16 * L1_PITCH); // [16][128]           8 KB
  float* hA  = xf + 16 * 128;                 // [16][24]
  float* hB  = hA + 16 * 24;                  // [16][24]
  float* nrm = hB + 16 * 24;                  // [2][16] double-buffered

  const int tid  = threadIdx.x;
  const int wv   = tid >> 5;    // wave 0..7
  const int lane = tid & 31;
  const int m16  = lane & 15;   // row within A / col within B,D
  const int hh   = lane >> 4;   // half-lane selector
  const int row0 = blockIdx.x * 16;

  const float tau = tau_ptr[0];
  const float sig = tau;

  __builtin_prefetch(W4, 0, 2);   // global_prefetch_b8
  __builtin_prefetch(Sg, 0, 2);

  // ---- stage S into LDS, both layouts (entries 0/1 -> f16 exact) ----
  for (int idx = tid; idx < N_COMBOS * N_STRUCTS; idx += 256) {
    int c = idx >> 9, s = idx & 511;
    _Float16 val = (_Float16)Sg[idx];
    Sc[c * SC_PITCH + s] = val;
    St[s * ST_PITCH + c] = val;
  }
  // ---- stage bids tile Xf[16][128]; it is also the LP rhs b ----
  for (int idx = tid; idx < 16 * 128; idx += 256) {
    int r = idx >> 7, c = idx & 127;
    xf[r * 128 + c] = X[(size_t)(row0 + r) * 128 + c];
  }
  if (tid < 32) nrm[tid] = 0.f;
  __syncthreads();

  // ---- MLP (tiny; plain VALU, hidden dim 20 is not WMMA-shaped) ----
  for (int idx = tid; idx < 16 * HID; idx += 256) {
    int r = idx / HID, j = idx % HID;
    float a = Bb1[j];
    for (int k = 0; k < 128; ++k) a += xf[r * 128 + k] * W1[k * HID + j];
    hA[r * 24 + j] = tanhf(a);
  }
  __syncthreads();
  for (int idx = tid; idx < 16 * HID; idx += 256) {
    int r = idx / HID, j = idx % HID;
    float a = Bb2[j];
    for (int k = 0; k < HID; ++k) a += hA[r * 24 + k] * W2[k * HID + j];
    hB[r * 24 + j] = tanhf(a);
  }
  __syncthreads();
  for (int idx = tid; idx < 16 * HID; idx += 256) {
    int r = idx / HID, j = idx % HID;
    float a = Bb3[j];
    for (int k = 0; k < HID; ++k) a += hB[r * 24 + k] * W3[k * HID + j];
    hA[r * 24 + j] = tanhf(a);
  }
  __syncthreads();

  // ---- z tile in registers (fragment layout), last MLP layer ----
  float zreg[4][8], xreg[4][8], lam2[4][8], lam1[8], breg[8];
#pragma unroll
  for (int t = 0; t < 4; ++t) {
    const int n = 64 * wv + 16 * t + m16;
#pragma unroll
    for (int r = 0; r < 8; ++r) zreg[t][r] = Bb4[n];
  }
  for (int j = 0; j < HID; ++j) {
    float w4v[4], hv[8];
#pragma unroll
    for (int t = 0; t < 4; ++t) w4v[t] = W4[j * N_STRUCTS + 64 * wv + 16 * t + m16];
#pragma unroll
    for (int r = 0; r < 8; ++r) hv[r] = hA[(r + 8 * hh) * 24 + j];
#pragma unroll
    for (int t = 0; t < 4; ++t)
#pragma unroll
      for (int r = 0; r < 8; ++r) zreg[t][r] += hv[r] * w4v[t];
  }

  // ---- PDHG init: x = max(z,0), xb = x, lam1 = lam2 = 0 ----
#pragma unroll
  for (int r = 0; r < 8; ++r) {
    lam1[r] = 0.f;
    breg[r] = xf[(r + 8 * hh) * 128 + 16 * wv + m16];
  }
#pragma unroll
  for (int t = 0; t < 4; ++t)
#pragma unroll
    for (int r = 0; r < 8; ++r) {
      float x0 = fmaxf(zreg[t][r], 0.f);
      xreg[t][r] = x0;
      lam2[t][r] = 0.f;
      xbh[(r + 8 * hh) * XB_PITCH + 64 * wv + 16 * t + m16] = (_Float16)x0;
    }
  __syncthreads();

  const v8f zero8 = {0.f, 0.f, 0.f, 0.f, 0.f, 0.f, 0.f, 0.f};

  for (int it = 0; it < PDHG_ITERS; ++it) {
    const int par = it & 1;

    // ---- u1 tile: (xb @ S^T) for this wave's combo tile, K=512 ----
    v8f acc = zero8;
    const int ao = m16 * XB_PITCH;
    const int bo = (16 * wv + m16) * SC_PITCH;  // B lane = S row (combo)
#pragma unroll 4
    for (int kk = 0; kk < 16; ++kk) {
      const int k0 = kk * 32 + 8 * hh;
      v16h a  = ld_frag(xbh + ao + k0, xbh + ao + k0 + 16);
      v16h bm = ld_frag(Sc + bo + k0, Sc + bo + k0 + 16);
      acc = __builtin_amdgcn_wmma_f32_16x16x32_f16(false, a, false, bm,
                                                   (short)0, acc, false, false);
    }
    // lam1 = max(lam1 + sig*(xb@S^T) - sig*b, 0); publish f16 for next GEMM
#pragma unroll
    for (int r = 0; r < 8; ++r) {
      float u1 = lam1[r] + sig * acc[r];
      float nl = fmaxf(u1 - sig * breg[r], 0.f);
      lam1[r] = nl;
      l1h[(r + 8 * hh) * L1_PITCH + 16 * wv + m16] = (_Float16)nl;
    }
    __syncthreads();  // lam1h complete; all u1 reads of xbh done

    // ---- p tiles: (lam1 @ S) for this wave's 4 structure tiles, K=128 ----
    v8f pacc[4];
#pragma unroll
    for (int t = 0; t < 4; ++t) pacc[t] = zero8;
    const int lo = m16 * L1_PITCH;
#pragma unroll
    for (int kk = 0; kk < 4; ++kk) {
      const int k0 = kk * 32 + 8 * hh;
      v16h a = ld_frag(l1h + lo + k0, l1h + lo + k0 + 16);
#pragma unroll
      for (int t = 0; t < 4; ++t) {
        const int s = 64 * wv + 16 * t + m16;  // B lane = S^T row (struct)
        v16h bm = ld_frag(St + s * ST_PITCH + k0, St + s * ST_PITCH + k0 + 16);
        pacc[t] = __builtin_amdgcn_wmma_f32_16x16x32_f16(
            false, a, false, bm, (short)0, pacc[t], false, false);
      }
    }

    // ---- elementwise prox + per-row norm partials ----
    float uu[4][8], pr[8];
#pragma unroll
    for (int r = 0; r < 8; ++r) pr[r] = 0.f;
#pragma unroll
    for (int t = 0; t < 4; ++t) {
      const int n = 64 * wv + 16 * t + m16;
#pragma unroll
      for (int r = 0; r < 8; ++r) {
        const int m = r + 8 * hh;
        float xbv = (float)xbh[m * XB_PITCH + n];
        float l2  = fminf(lam2[t][r] + sig * xbv, 0.f);
        lam2[t][r] = l2;
        float vv = xreg[t][r] - tau * (pacc[t][r] + l2);
        float u  = vv + tau - zreg[t][r];   // w == 1
        uu[t][r] = u;
        pr[r] += u * u;
      }
    }
    if (tid < 16) nrm[(par ^ 1) * 16 + tid] = 0.f;  // prep other buffer
    // reduce partials across the 16-lane half, then one ds_add_f32 per row
#pragma unroll
    for (int r = 0; r < 8; ++r) {
      float s = pr[r];
      s += __shfl_xor(s, 1, 32);
      s += __shfl_xor(s, 2, 32);
      s += __shfl_xor(s, 4, 32);
      s += __shfl_xor(s, 8, 32);
      if (m16 == 0) atomicAdd(&nrm[par * 16 + r + 8 * hh], s);
    }
    __syncthreads();  // row norms complete

    float scl[8];
#pragma unroll
    for (int r = 0; r < 8; ++r) {
      float nr = sqrtf(nrm[par * 16 + r + 8 * hh]);
      scl[r] = fmaxf(0.f, 1.f - tau * CONTROLF / fmaxf(nr, 1e-12f));
    }
#pragma unroll
    for (int t = 0; t < 4; ++t) {
      const int n = 64 * wv + 16 * t + m16;
#pragma unroll
      for (int r = 0; r < 8; ++r) {
        float xn = zreg[t][r] + scl[r] * uu[t][r];
        float xb = 2.f * xn - xreg[t][r];
        xreg[t][r] = xn;
        xbh[(r + 8 * hh) * XB_PITCH + n] = (_Float16)xb;
      }
    }
    __syncthreads();  // xb published for next iteration's u1 GEMM
  }

  // ---- write allocation x ----
#pragma unroll
  for (int t = 0; t < 4; ++t) {
    const int n = 64 * wv + 16 * t + m16;
#pragma unroll
    for (int r = 0; r < 8; ++r)
      out[(size_t)(row0 + r + 8 * hh) * N_STRUCTS + n] = xreg[t][r];
  }
}

// ---------------------------------------------------------------------------
extern "C" void kernel_launch(void* const* d_in, const int* in_sizes, int n_in,
                              void* d_out, int out_size, void* d_ws,
                              size_t ws_size, hipStream_t stream) {
  (void)in_sizes; (void)n_in; (void)out_size; (void)ws_size;
  const float* X   = (const float*)d_in[0];
  const float* W1  = (const float*)d_in[1];
  const float* Bb1 = (const float*)d_in[2];
  const float* W2  = (const float*)d_in[3];
  const float* Bb2 = (const float*)d_in[4];
  const float* W3  = (const float*)d_in[5];
  const float* Bb3 = (const float*)d_in[6];
  const float* W4  = (const float*)d_in[7];
  const float* Bb4 = (const float*)d_in[8];
  const float* Sg  = (const float*)d_in[9];
  float* tau = (float*)d_ws;   // one float of scratch
  float* out = (float*)d_out;

  const size_t spec_smem =
      (size_t)(N_COMBOS * N_STRUCTS + N_STRUCTS + N_COMBOS + 256) * sizeof(float);
  const size_t pdhg_smem =
      (size_t)(N_COMBOS * SC_PITCH + N_STRUCTS * ST_PITCH + 16 * XB_PITCH +
               16 * L1_PITCH) * sizeof(_Float16) +
      (size_t)(16 * 128 + 16 * 24 + 16 * 24 + 32) * sizeof(float);  // ~305 KB < 320 KB WGP LDS

  hipFuncSetAttribute((const void*)spectral_kernel,
                      hipFuncAttributeMaxDynamicSharedMemorySize, (int)spec_smem);
  hipFuncSetAttribute((const void*)pdhg_kernel,
                      hipFuncAttributeMaxDynamicSharedMemorySize, (int)pdhg_smem);

  spectral_kernel<<<1, 256, spec_smem, stream>>>(Sg, tau);
  pdhg_kernel<<<BATCHSZ / 16, 256, pdhg_smem, stream>>>(
      X, W1, Bb1, W2, Bb2, W3, Bb3, W4, Bb4, Sg, tau, out);
}
